// GraphMatching_78726750535698
// MI455X (gfx1250) — matrix operand order repaired and checked by reading, not compiled, not run on hardware
//
#include <hip/hip_runtime.h>
#include <hip/hip_bf16.h>

// ---------------- problem constants ----------------
constexpr int kQG = 64, kSG = 16, kG = 256, kF = 512, kH = 256;
constexpr int kEQ = 262144, kES = 65536;
constexpr int kQN = kQG * kG;   // 16384
constexpr int kSN = kSG * kG;   // 4096
constexpr int kGH = kG * kH;    // 65536 elems per 256x256 matrix

typedef __bf16 bf16_t;
typedef __attribute__((ext_vector_type(16))) __bf16 v16bf;
typedef __attribute__((ext_vector_type(8)))  float  v8f;

static __device__ inline v8f v8f_zero() {
  v8f z;
#pragma unroll
  for (int i = 0; i < 8; ++i) z[i] = 0.0f;
  return z;
}

static __device__ inline v8f wmma_bf16(v16bf a, v16bf b, v8f c) {
  // D = A(16x32 bf16) x B(32x16 bf16) + C(16x16 f32)
  return __builtin_amdgcn_wmma_f32_16x16x32_bf16(false, a, false, b, (short)0, c,
                                                 false, false);
}

// ---- fragment loaders (per CDNA5 ISA 7.12.2 VGPR layouts, wave32) ----
// A 16x32 bf16: lane = {m=lane&15 -> row, h=lane>>4}; 16 bf16 per lane:
//   elems 0..7  -> K = k0 + h*8 + {0..7}
//   elems 8..15 -> K = k0 + 16 + h*8 + {0..7}
static __device__ inline v16bf load_a_bf16(const bf16_t* A, int ld, int row0,
                                           int k0, int lane) {
  int m = lane & 15, h = lane >> 4;
  const bf16_t* p = A + (size_t)(row0 + m) * ld + k0 + h * 8;
  v16bf r;
#pragma unroll
  for (int i = 0; i < 8; ++i) { r[i] = p[i]; r[8 + i] = p[16 + i]; }
  return r;
}

static __device__ inline v16bf load_a_f32(const float* A, int ld, int row0,
                                          int k0, int lane) {
  int m = lane & 15, h = lane >> 4;
  const float* p = A + (size_t)(row0 + m) * ld + k0 + h * 8;
  v16bf r;
#pragma unroll
  for (int i = 0; i < 8; ++i) {
    r[i] = (bf16_t)p[i];
    r[8 + i] = (bf16_t)p[16 + i];
  }
  return r;
}

static __device__ inline v16bf load_a_f32_relu_scale(const float* A, int ld,
                                                     int row0, int k0, int lane,
                                                     float sc) {
  int m = lane & 15, h = lane >> 4;
  const float* p = A + (size_t)(row0 + m) * ld + k0 + h * 8;
  v16bf r;
#pragma unroll
  for (int i = 0; i < 8; ++i) {
    r[i]     = (bf16_t)fmaxf(p[i] * sc, 0.0f);
    r[8 + i] = (bf16_t)fmaxf(p[16 + i] * sc, 0.0f);
  }
  return r;
}

// B 32x16 bf16: lane holds col n0 + (lane&15); elems 0..7 -> K = k0 + h*8 + i,
//   elems 8..15 -> K = k0 + 16 + h*8 + i (h = lane>>4).
static __device__ inline v16bf load_b_bf16(const bf16_t* B, int ld, int k0,
                                           int n0, int lane) {
  const bf16_t* p = B + (size_t)(k0 + (lane >> 4) * 8) * ld + n0 + (lane & 15);
  v16bf r;
#pragma unroll
  for (int i = 0; i < 8; ++i) { r[i] = p[(size_t)i * ld]; r[8 + i] = p[(size_t)(16 + i) * ld]; }
  return r;
}

// Row softmax over a 16x256 panel held as 16 accumulator tiles.
// Row = r + 8*h (h = lane>>4); elements spread over tiles t and lanes of the
// 16-lane half -> reduce with __shfl_xor width 16.
static __device__ inline void softmax16(v8f acc[16]) {
#pragma unroll
  for (int r = 0; r < 8; ++r) {
    float mx = -3.4e38f;
#pragma unroll
    for (int t = 0; t < 16; ++t) mx = fmaxf(mx, acc[t][r]);
#pragma unroll
    for (int o = 1; o < 16; o <<= 1) mx = fmaxf(mx, __shfl_xor(mx, o, 16));
    float s = 0.0f;
#pragma unroll
    for (int t = 0; t < 16; ++t) {
      float e = __expf(acc[t][r] - mx);
      acc[t][r] = e;
      s += e;
    }
#pragma unroll
    for (int o = 1; o < 16; o <<= 1) s += __shfl_xor(s, o, 16);
    float inv = 1.0f / s;
#pragma unroll
    for (int t = 0; t < 16; ++t) acc[t][r] *= inv;
  }
}

// Dump a 16x256 f32 panel (16 tiles) as bf16 into per-wave LDS panel [16][256].
static __device__ inline void store_panel(bf16_t* panel, const v8f acc[16],
                                          int m, int h) {
#pragma unroll
  for (int t = 0; t < 16; ++t)
#pragma unroll
    for (int r = 0; r < 8; ++r)
      panel[(size_t)(r + 8 * h) * 256 + t * 16 + m] = (bf16_t)acc[t][r];
}

// ---------------- small utility kernels ----------------
__global__ void fill_f32_kernel(float* p, float v, int n) {
  int i = blockIdx.x * blockDim.x + threadIdx.x;
  if (i < n) p[i] = v;
}

__global__ void pack_bf16_kernel(const float* __restrict__ src,
                                 bf16_t* __restrict__ dst, size_t n) {
  size_t i = (size_t)blockIdx.x * blockDim.x + threadIdx.x;
  if (i < n) dst[i] = (bf16_t)src[i];
}

// Batched 256x256 bf16 transpose (LDS-tiled 32x32). grid.x = nmat * 64.
// Coalesced 64B rows on both load and store; padded LDS avoids conflicts.
__global__ void __launch_bounds__(256)
transpose_kernel(const bf16_t* __restrict__ src, bf16_t* __restrict__ dst) {
  __shared__ bf16_t tile[32][33];
  int b = blockIdx.x;
  int mat = b >> 6;
  int t = b & 63;
  int tx = t & 7, ty = t >> 3;  // 32-wide tile coords
  const bf16_t* S = src + (size_t)mat * kGH;
  bf16_t* D = dst + (size_t)mat * kGH;
  int x = threadIdx.x & 31;
  int y8 = threadIdx.x >> 5;
#pragma unroll
  for (int i = 0; i < 4; ++i) {
    int y = y8 + i * 8;
    tile[y][x] = S[(size_t)(ty * 32 + y) * kH + tx * 32 + x];
  }
  __syncthreads();
#pragma unroll
  for (int i = 0; i < 4; ++i) {
    int y = y8 + i * 8;
    D[(size_t)(tx * 32 + y) * kH + ty * 32 + x] = tile[x][y];
  }
}

// pooled[dst] += x[src]; deg[dst] += 1     (one block per edge)
__global__ void scatter_kernel(const float* __restrict__ x,
                               const int* __restrict__ ei, int nE, int nMask,
                               float* __restrict__ pooled,
                               float* __restrict__ deg) {
  int e = blockIdx.x;
  if (e >= nE) return;
  int src = ei[e] & nMask;
  int dst = ei[nE + e] & nMask;
  const float* xs = x + (size_t)src * kF;
  float* pd = pooled + (size_t)dst * kF;
  for (int f = threadIdx.x; f < kF; f += blockDim.x) atomicAdd(&pd[f], xs[f]);
  if (threadIdx.x == 0) atomicAdd(&deg[dst], 1.0f);
}

// out = relu((pooled @ Wgc + bgc) / deg), K = 512. 4 waves/block, one
// 16-row stripe per wave, A-fragments cached in registers.
__global__ void __launch_bounds__(128)
gcn_gemm_kernel(const float* __restrict__ pooled, const bf16_t* __restrict__ Wgc,
                const float* __restrict__ bgc, const float* __restrict__ deg,
                float* __restrict__ outF, bf16_t* __restrict__ outB) {
  int lane = threadIdx.x & 31, m = lane & 15, h = lane >> 4;
  int stripe = blockIdx.x * 4 + (threadIdx.x >> 5);
  int row0 = stripe * 16;
  v16bf af[16];
#pragma unroll
  for (int ks = 0; ks < 16; ++ks) af[ks] = load_a_f32(pooled, kF, row0, ks * 32, lane);
  float invd[8];
#pragma unroll
  for (int r = 0; r < 8; ++r) invd[r] = 1.0f / deg[row0 + r + 8 * h];
  for (int nt = 0; nt < 16; ++nt) {
    v8f acc = v8f_zero();
#pragma unroll
    for (int ks = 0; ks < 16; ++ks)
      acc = wmma_bf16(af[ks], load_b_bf16(Wgc, kH, ks * 32, nt * 16, lane), acc);
    float bn = bgc[nt * 16 + m];
#pragma unroll
    for (int r = 0; r < 8; ++r) {
      float v = fmaxf((acc[r] + bn) * invd[r], 0.0f);
      size_t idx = (size_t)(row0 + r + 8 * h) * kH + nt * 16 + m;
      if (outF) outF[idx] = v;
      if (outB) outB[idx] = (bf16_t)v;
    }
  }
}

// qW[c] = q_bf[c] @ Wmatch   (batched; one 16x16 tile per wave)
__global__ void __launch_bounds__(128)
gemm_qw_kernel(const bf16_t* __restrict__ q, const bf16_t* __restrict__ Wm,
               bf16_t* __restrict__ qW) {
  int lane = threadIdx.x & 31, m = lane & 15, h = lane >> 4;
  int tid = blockIdx.x * 4 + (threadIdx.x >> 5);
  int nt = tid & 15, stripe = (tid >> 4) & 15, c = tid >> 8;
  const bf16_t* A = q + (size_t)c * kGH;
  v8f acc = v8f_zero();
#pragma unroll
  for (int ks = 0; ks < 8; ++ks) {
    v16bf a = load_a_bf16(A, kH, stripe * 16, ks * 32, lane);
    v16bf b = load_b_bf16(Wm, kH, ks * 32, nt * 16, lane);
    acc = wmma_bf16(a, b, acc);
  }
  bf16_t* C = qW + (size_t)c * kGH;
#pragma unroll
  for (int r = 0; r < 8; ++r)
    C[(size_t)(stripe * 16 + r + 8 * h) * kH + nt * 16 + m] = (bf16_t)acc[r];
}

// Fused match kernel: one block = (query c, support s, quarter part);
// 4 waves, each wave owns a 16-row stripe (row0) of the 256x256 matrices.
// All B operands are contiguous (explicit yT / qT transposes).
//  Part A:  P = softmax(qW @ yT); msum[c] += P @ y    (atomic over supports)
//  Part B:  yW = y @ Wm; P2 = softmax(yW @ qT); ms = P2 @ q;
//           Ys_out[c,s] = [ms | y] @ Wagg + bagg      (skipped on last layer)
__global__ void __launch_bounds__(128)
match_kernel(const bf16_t* __restrict__ qW, const bf16_t* __restrict__ qbf,
             const bf16_t* __restrict__ qT, const bf16_t* __restrict__ ybase,
             long long ycs, long long yss, const bf16_t* __restrict__ ytbase,
             long long ytcs, long long ytss, const bf16_t* __restrict__ Wm,
             const bf16_t* __restrict__ Wagg, const float* __restrict__ bagg,
             float* __restrict__ msum, bf16_t* __restrict__ yout, int doAgg) {
  extern __shared__ char smraw[];
  int lane = threadIdx.x & 31, m = lane & 15, h = lane >> 4;
  int wave = threadIdx.x >> 5;
  int bid = blockIdx.x;
  int part = bid & 3, s = (bid >> 2) & 15, c = bid >> 6;
  int row0 = part * 64 + wave * 16;
  const bf16_t* y   = ybase + (long long)c * ycs + (long long)s * yss;
  const bf16_t* yt  = ytbase + (long long)c * ytcs + (long long)s * ytss;
  const bf16_t* qWc = qW + (size_t)c * kGH;
  const bf16_t* qc  = qbf + (size_t)c * kGH;
  const bf16_t* qTc = qT + (size_t)c * kGH;
  bf16_t* panel = (bf16_t*)smraw + wave * (16 * 256);

  __builtin_prefetch(y + (size_t)row0 * kH, 0, 3);   // global_prefetch_b8
  __builtin_prefetch(yt, 0, 3);

  v8f acc[16];

  // ---------- Part A ----------
#pragma unroll
  for (int t = 0; t < 16; ++t) acc[t] = v8f_zero();
#pragma unroll
  for (int ks = 0; ks < 8; ++ks) {
    v16bf a = load_a_bf16(qWc, kH, row0, ks * 32, lane);
#pragma unroll
    for (int t = 0; t < 16; ++t)
      acc[t] = wmma_bf16(a, load_b_bf16(yt, kH, ks * 32, t * 16, lane), acc[t]);
  }
  softmax16(acc);
  store_panel(panel, acc, m, h);
#pragma unroll
  for (int t = 0; t < 16; ++t) acc[t] = v8f_zero();
#pragma unroll
  for (int ks = 0; ks < 8; ++ks) {
    v16bf a = load_a_bf16(panel, kH, 0, ks * 32, lane);
#pragma unroll
    for (int t = 0; t < 16; ++t)
      acc[t] = wmma_bf16(a, load_b_bf16(y, kH, ks * 32, t * 16, lane), acc[t]);
  }
  float* mc = msum + (size_t)c * kGH;
#pragma unroll
  for (int t = 0; t < 16; ++t)
#pragma unroll
    for (int r = 0; r < 8; ++r)
      atomicAdd(&mc[(size_t)(row0 + r + 8 * h) * kH + t * 16 + m], acc[t][r]);

  // ---------- Part B ----------
  if (doAgg) {
    // yW = y @ Wm -> panel
#pragma unroll
    for (int t = 0; t < 16; ++t) acc[t] = v8f_zero();
#pragma unroll
    for (int ks = 0; ks < 8; ++ks) {
      v16bf a = load_a_bf16(y, kH, row0, ks * 32, lane);
#pragma unroll
      for (int t = 0; t < 16; ++t)
        acc[t] = wmma_bf16(a, load_b_bf16(Wm, kH, ks * 32, t * 16, lane), acc[t]);
    }
    store_panel(panel, acc, m, h);
    // logits2 = yW @ qT
#pragma unroll
    for (int t = 0; t < 16; ++t) acc[t] = v8f_zero();
#pragma unroll
    for (int ks = 0; ks < 8; ++ks) {
      v16bf a = load_a_bf16(panel, kH, 0, ks * 32, lane);
#pragma unroll
      for (int t = 0; t < 16; ++t)
        acc[t] = wmma_bf16(a, load_b_bf16(qTc, kH, ks * 32, t * 16, lane), acc[t]);
    }
    softmax16(acc);
    store_panel(panel, acc, m, h);  // P2
    // ms = P2 @ q
#pragma unroll
    for (int t = 0; t < 16; ++t) acc[t] = v8f_zero();
#pragma unroll
    for (int ks = 0; ks < 8; ++ks) {
      v16bf a = load_a_bf16(panel, kH, 0, ks * 32, lane);
#pragma unroll
      for (int t = 0; t < 16; ++t)
        acc[t] = wmma_bf16(a, load_b_bf16(qc, kH, ks * 32, t * 16, lane), acc[t]);
    }
    store_panel(panel, acc, m, h);  // ms as bf16
    // out = [ms | y] @ Wagg + bagg
#pragma unroll
    for (int t = 0; t < 16; ++t) acc[t] = v8f_zero();
#pragma unroll
    for (int ks = 0; ks < 16; ++ks) {
      int k0 = ks * 32;
      v16bf a = (ks < 8) ? load_a_bf16(panel, kH, 0, k0, lane)
                         : load_a_bf16(y, kH, row0, k0 - kH, lane);
#pragma unroll
      for (int t = 0; t < 16; ++t)
        acc[t] = wmma_bf16(a, load_b_bf16(Wagg, kH, k0, t * 16, lane), acc[t]);
    }
    bf16_t* yo = yout + ((size_t)c * kSG + s) * kGH;
#pragma unroll
    for (int t = 0; t < 16; ++t) {
      float bn = bagg[t * 16 + m];
#pragma unroll
      for (int r = 0; r < 8; ++r)
        yo[(size_t)(row0 + r + 8 * h) * kH + t * 16 + m] = (bf16_t)(acc[t][r] + bn);
    }
  }
}

// q[c] = [relu(msum[c]/16) | q[c]] @ Wagg + bagg   (in-place; A cached first)
__global__ void __launch_bounds__(128)
qupdate_kernel(const float* __restrict__ msum, float* __restrict__ qx,
               const bf16_t* __restrict__ Wagg, const float* __restrict__ bagg,
               int c0) {
  int lane = threadIdx.x & 31, m = lane & 15, h = lane >> 4;
  int stripe = blockIdx.x * 4 + (threadIdx.x >> 5);
  int c = stripe >> 4;
  int row0 = (stripe & 15) * 16;
  const float* mc = msum + (size_t)c * kGH;
  float* qc = qx + (size_t)(c0 + c) * kGH;
  v16bf af[16];
#pragma unroll
  for (int ks = 0; ks < 8; ++ks)
    af[ks] = load_a_f32_relu_scale(mc, kH, row0, ks * 32, lane, 1.0f / 16.0f);
#pragma unroll
  for (int ks = 8; ks < 16; ++ks)
    af[ks] = load_a_f32(qc, kH, row0, (ks - 8) * 32, lane);
  for (int nt = 0; nt < 16; ++nt) {
    v8f acc = v8f_zero();
#pragma unroll
    for (int ks = 0; ks < 16; ++ks)
      acc = wmma_bf16(af[ks], load_b_bf16(Wagg, kH, ks * 32, nt * 16, lane), acc);
    float bn = bagg[nt * 16 + m];
#pragma unroll
    for (int r = 0; r < 8; ++r)
      qc[(size_t)(row0 + r + 8 * h) * kH + nt * 16 + m] = acc[r] + bn;
  }
}

// out[c] = mean(q @ Wfin + bfin) over rows
__global__ void __launch_bounds__(256)
final_kernel(const float* __restrict__ qx, const float* __restrict__ Wfin,
             const float* __restrict__ bfin, float* __restrict__ out, int c0) {
  __shared__ float red[256];
  int c = blockIdx.x, hh = threadIdx.x;
  const float* q = qx + (size_t)(c0 + c) * kGH;
  float s = 0.0f;
  for (int i = 0; i < kG; ++i) s += q[(size_t)i * kH + hh];
  red[hh] = s * Wfin[hh];
  __syncthreads();
  for (int o = 128; o > 0; o >>= 1) {
    if (hh < o) red[hh] += red[hh + o];
    __syncthreads();
  }
  if (hh == 0) out[c0 + c] = red[0] / (float)kG + bfin[0];
}

// ---------------- host orchestration ----------------
extern "C" void kernel_launch(void* const* d_in, const int* in_sizes, int n_in,
                              void* d_out, int out_size, void* d_ws,
                              size_t ws_size, hipStream_t stream) {
  (void)in_sizes; (void)n_in; (void)out_size;
  const float* xq   = (const float*)d_in[0];
  const float* xs   = (const float*)d_in[1];
  const float* Wgc  = (const float*)d_in[2];
  const float* bgc  = (const float*)d_in[3];
  const float* Wm   = (const float*)d_in[4];
  const float* Wagg = (const float*)d_in[5];
  const float* bagg = (const float*)d_in[6];
  const float* Wfin = (const float*)d_in[7];
  const float* bfin = (const float*)d_in[8];
  const int*   eq   = (const int*)d_in[9];
  const int*   es   = (const int*)d_in[10];
  float* out = (float*)d_out;

  char* ws = (char*)d_ws;
  size_t off = 0;
  auto alloc = [&](size_t bytes) -> char* {
    char* p = ws + off;
    off = (off + bytes + 255) & ~(size_t)255;
    return p;
  };
  float*  qx   = (float*)alloc((size_t)kQN * kH * 4);
  bf16_t* sxb  = (bf16_t*)alloc((size_t)kSN * kH * 2);
  bf16_t* sxT  = (bf16_t*)alloc((size_t)kSN * kH * 2);   // per-support transposes
  bf16_t* WgcB = (bf16_t*)alloc((size_t)kF * kH * 2);
  bf16_t* WmB  = (bf16_t*)alloc((size_t)kH * kH * 2);
  bf16_t* WaB  = (bf16_t*)alloc((size_t)2 * kH * kH * 2);
  float*  degq = (float*)alloc((size_t)kQN * 4);
  float*  degs = (float*)alloc((size_t)kSN * 4);
  size_t fixedEnd = off;
  char* rx = ws + fixedEnd;
  size_t rxAvail = (ws_size > fixedEnd) ? ws_size - fixedEnd : 0;

  // ---- phase 1: GCN (region X used as pooled buffers) ----
  float* pooledQ = (float*)rx;
  float* pooledS = (float*)(rx + (size_t)kQN * kF * 4);
  hipMemcpyAsync(pooledQ, xq, (size_t)kQN * kF * 4, hipMemcpyDeviceToDevice, stream);
  hipMemcpyAsync(pooledS, xs, (size_t)kSN * kF * 4, hipMemcpyDeviceToDevice, stream);
  fill_f32_kernel<<<(kQN + 255) / 256, 256, 0, stream>>>(degq, 1.0f, kQN);
  fill_f32_kernel<<<(kSN + 255) / 256, 256, 0, stream>>>(degs, 1.0f, kSN);
  pack_bf16_kernel<<<(kF * kH + 255) / 256, 256, 0, stream>>>(Wgc, WgcB, (size_t)kF * kH);
  pack_bf16_kernel<<<(kH * kH + 255) / 256, 256, 0, stream>>>(Wm, WmB, (size_t)kH * kH);
  pack_bf16_kernel<<<(2 * kH * kH + 255) / 256, 256, 0, stream>>>(Wagg, WaB, (size_t)2 * kH * kH);
  scatter_kernel<<<kEQ, 128, 0, stream>>>(xq, eq, kEQ, kQN - 1, pooledQ, degq);
  scatter_kernel<<<kES, 128, 0, stream>>>(xs, es, kES, kSN - 1, pooledS, degs);
  gcn_gemm_kernel<<<kQN / 64, 128, 0, stream>>>(pooledQ, WgcB, bgc, degq, qx, nullptr);
  gcn_gemm_kernel<<<kSN / 64, 128, 0, stream>>>(pooledS, WgcB, bgc, degs, nullptr, sxb);
  transpose_kernel<<<kSG * 64, 256, 0, stream>>>(sxb, sxT);

  // ---- phase 2: matching, chunked over queries to fit ws ----
  const size_t perQ = (size_t)kGH * 2          /* q_bf  */
                    + (size_t)kGH * 2          /* qW_bf */
                    + (size_t)kGH * 2          /* qT    */
                    + (size_t)kGH * 4          /* msum  */
                    + (size_t)kSG * kGH * 2    /* Ys    */
                    + (size_t)kSG * kGH * 2;   /* YsT   */
  int nc = (int)(rxAvail / perQ);
  if (nc < 1) nc = 1;
  if (nc > kQG) nc = kQG;

  bf16_t* qbf  = (bf16_t*)rx;
  bf16_t* qWb  = qbf + (size_t)nc * kGH;
  bf16_t* qTb  = qWb + (size_t)nc * kGH;
  float*  msum = (float*)(qTb + (size_t)nc * kGH);
  bf16_t* Ys   = (bf16_t*)(msum + (size_t)nc * kGH);
  bf16_t* YsT  = Ys + (size_t)nc * kSG * kGH;

  for (int c0 = 0; c0 < kQG; c0 += nc) {
    int n = (kQG - c0 < nc) ? (kQG - c0) : nc;
    for (int layer = 0; layer < 2; ++layer) {
      pack_bf16_kernel<<<(n * kGH + 255) / 256, 256, 0, stream>>>(
          qx + (size_t)c0 * kGH, qbf, (size_t)n * kGH);
      transpose_kernel<<<n * 64, 256, 0, stream>>>(qbf, qTb);
      gemm_qw_kernel<<<n * 64, 128, 0, stream>>>(qbf, WmB, qWb);
      hipMemsetAsync(msum, 0, (size_t)n * kGH * 4, stream);
      int doAgg = (layer == 0);  // last layer's Ys is never consumed
      const bf16_t* yb;
      const bf16_t* ytb;
      long long ycs, yss;
      if (layer == 0) {
        yb = sxb;  ytb = sxT;  ycs = 0;  yss = kGH;
      } else {
        transpose_kernel<<<n * kSG * 64, 256, 0, stream>>>(Ys, YsT);
        yb = Ys;   ytb = YsT;  ycs = (long long)kSG * kGH;  yss = kGH;
      }
      match_kernel<<<n * kSG * 4, 128, 4 * 16 * 256 * 2, stream>>>(
          qWb, qbf, qTb, yb, ycs, yss, ytb, ycs, yss, WmB, WaB, bagg, msum,
          doAgg ? Ys : nullptr, doAgg);
      qupdate_kernel<<<n * 4, 128, 0, stream>>>(msum, qx, WaB, bagg, c0);
    }
    final_kernel<<<n, 256, 0, stream>>>(qx, Wfin, bfin, out, c0);
  }
}
